// DynamicKernelConv2D_50723563765872
// MI455X (gfx1250) — matrix operand order repaired and checked by reading, not compiled, not run on hardware
//
#include <hip/hip_runtime.h>
#include <hip/hip_bf16.h>

typedef float  v8f   __attribute__((ext_vector_type(8)));
typedef __bf16 v8bf  __attribute__((ext_vector_type(8)));
typedef __bf16 v16bf __attribute__((ext_vector_type(16)));

#define BATCH 16
#define IC 128
#define OC 128
#define HW 64
#define LDK 40   // LDS row stride in bf16 elems (32 K + pad, 80B keeps 16B align)

// gfx1250 async global->LDS path (gated: falls back to plain copies if the
// toolchain doesn't declare the builtins). Param type per hipcc diagnostic:
// '__attribute__((__vector_size__(4*sizeof(int)))) int __device__ *'
#if __has_builtin(__builtin_amdgcn_global_load_async_to_lds_b128) && \
    __has_builtin(__builtin_amdgcn_s_wait_asynccnt)
#define USE_ASYNC_LDS 1
typedef int v4i __attribute__((vector_size(4 * sizeof(int))));
#define AS1P(p) ((__attribute__((address_space(1))) v4i*)(uintptr_t)(p))
#define AS3P(p) ((__attribute__((address_space(3))) v4i*)(uintptr_t)(p))
#endif

// ---------------- global average pool: x(B,C,64,64) -> pooled(B,C) ----------
__global__ __launch_bounds__(256) void pool_kernel(const float* __restrict__ x,
                                                   float* __restrict__ pooled) {
    __shared__ float red[256];
    const int bc = blockIdx.x;                 // b*128 + c
    const float* p = x + (size_t)bc * (HW * HW);
    float s = 0.0f;
    for (int i = threadIdx.x; i < HW * HW; i += 256) s += p[i];
    red[threadIdx.x] = s;
    __syncthreads();
    for (int st = 128; st > 0; st >>= 1) {
        if (threadIdx.x < st) red[threadIdx.x] += red[threadIdx.x + st];
        __syncthreads();
    }
    if (threadIdx.x == 0) pooled[bc] = red[0] * (1.0f / (HW * HW));
}

// ---- params = pooled @ pred_w.T + pred_b ; mask(b,ky,kx) gaussian ----------
__global__ __launch_bounds__(256) void mask_kernel(const float* __restrict__ pooled,
                                                   const float* __restrict__ pw,
                                                   const float* __restrict__ pb,
                                                   float* __restrict__ maskbuf) {
    __shared__ float par[BATCH * 3];
    const int t = threadIdx.x;
    if (t < BATCH * 3) {
        const int b = t / 3, j = t % 3;
        float s = pb[j];
        for (int c = 0; c < IC; ++c) s += pooled[b * IC + c] * pw[j * IC + c];
        par[t] = s;
    }
    __syncthreads();
    for (int e = t; e < BATCH * 49; e += 256) {
        const int b = e / 49, pos = e % 49, ky = pos / 7, kx = pos % 7;
        const float p0 = par[b * 3 + 0], p1 = par[b * 3 + 1], p2 = par[b * 3 + 2];
        const float ah = 1.0f / (1.0f + expf(-p0));             // sigmoid
        const float aw = 1.0f / (1.0f + expf(-p1));
        const float sp = (p2 > 20.0f) ? p2 : log1pf(expf(p2));  // softplus
        const float sg = sp + 0.001f;
        const float yg = (float)ky * (1.0f / 6.0f);             // linspace(0,1,7)
        const float xg = (float)kx * (1.0f / 6.0f);
        const float dx = xg - aw, dy = yg - ah;
        maskbuf[e] = expf(-(dx * dx + dy * dy) / (2.0f * sg * sg));
    }
}

// ---- one-shot: split x(f32) into bf16 hi/lo --------------------------------
__global__ __launch_bounds__(256) void split_x_kernel(const float* __restrict__ x,
                                                      __bf16* __restrict__ xHi,
                                                      __bf16* __restrict__ xLo) {
    const size_t i = (size_t)blockIdx.x * 256 + threadIdx.x;
    const float v = x[i];
    const __bf16 hi = (__bf16)v;
    xHi[i] = hi;
    xLo[i] = (__bf16)(v - (float)hi);
}

// ---- one-shot: rearrange+split weights into staging-order bf16 tiles -------
// wT[koff][chunk][o][k] = w[o][chunk*32+k][koff], so each (koff,chunk) A-tile
// is a fully contiguous 4096-elem (8KB) block.
__global__ __launch_bounds__(256) void split_w_kernel(const float* __restrict__ w,
                                                      __bf16* __restrict__ wtHi,
                                                      __bf16* __restrict__ wtLo) {
    const int i = blockIdx.x * 256 + threadIdx.x;     // 49*4*128*32 = 802816
    const int k     = i & 31;
    const int o     = (i >> 5) & 127;
    const int chunk = (i >> 12) & 3;
    const int koff  = i >> 14;                        // 0..48
    const float v = w[(size_t)(o * IC + chunk * 32 + k) * 49 + koff];
    const __bf16 hi = (__bf16)v;
    wtHi[i] = hi;
    wtLo[i] = (__bf16)(v - (float)hi);
}

// --------- implicit-GEMM conv via V_WMMA_F32_16X16X32_BF16 (hi/lo split) ----
// grid = (64 rows, 16 images), block = 256 threads = 8 waves.
// Mask is deferred: per (ky,kx) slab the unscaled GEMM accumulates in tmp,
// then acc += mval*tmp in f32. A-tiles stream to LDS via async-to-LDS.
__global__ __launch_bounds__(256) void conv_wmma_kernel(
    const __bf16* __restrict__ wtHi, const __bf16* __restrict__ wtLo,
    const __bf16* __restrict__ xHi,  const __bf16* __restrict__ xLo,
    const float* __restrict__ bias,  const float* __restrict__ maskbuf,
    float* __restrict__ conv, float* __restrict__ gsum, float* __restrict__ gsq) {
    __shared__ __attribute__((aligned(16))) __bf16 ldsAhi[OC * LDK];  // 128 x 32
    __shared__ __attribute__((aligned(16))) __bf16 ldsAlo[OC * LDK];
    __shared__ __attribute__((aligned(16))) __bf16 ldsBhi[HW * LDK];  // 64(n) x 32(k), transposed
    __shared__ __attribute__((aligned(16))) __bf16 ldsBlo[HW * LDK];

    const int y    = blockIdx.x;       // output row == pixel tile of 64
    const int b    = blockIdx.y;
    const int t    = threadIdx.x;
    const int wave = t >> 5;
    const int lane = t & 31;
    const int m16  = lane & 15;
    const int half = lane >> 4;
    const int mo   = wave * 16;        // this wave's 16 output channels

    v8f acc[4] = {};                   // final f32 accum, 4 subtiles of 16x16

    // A staging: thread t moves 16 contiguous bf16 (2 x 16B) per tile
    const int aoRow = (t >> 1) * LDK + (t & 1) * 16;
    const int aSrc  = t * 16;

    const __bf16* xb = xHi + (size_t)b * IC * HW * HW;
    const __bf16* xl = xLo + (size_t)b * IC * HW * HW;

    for (int ky = 0; ky < 7; ++ky) {
        const int iy = y + ky - 3;
        for (int kx = 0; kx < 7; ++kx) {
            const int koff = ky * 7 + kx;
            const float mval = maskbuf[b * 49 + koff];   // uniform -> scalar load
            v8f tmp[4] = {};                             // unscaled slab partials
#pragma unroll 1
            for (int chunk = 0; chunk < 4; ++chunk) {
                const int cb = chunk * 32;
                const __bf16* sH = wtHi + (size_t)(koff * 4 + chunk) * 4096;
                const __bf16* sL = wtLo + (size_t)(koff * 4 + chunk) * 4096;
                // prefetch next (koff,chunk) tile's weights (global_prefetch_b8)
                __builtin_prefetch(sH + 4096 + aSrc, 0, 1);
#ifdef USE_ASYNC_LDS
                // stage A: async global->LDS b128, no VGPR round-trip
                __builtin_amdgcn_global_load_async_to_lds_b128(
                    AS1P(&sH[aSrc]),     AS3P(&ldsAhi[aoRow]),     0, 0);
                __builtin_amdgcn_global_load_async_to_lds_b128(
                    AS1P(&sH[aSrc + 8]), AS3P(&ldsAhi[aoRow + 8]), 0, 0);
                __builtin_amdgcn_global_load_async_to_lds_b128(
                    AS1P(&sL[aSrc]),     AS3P(&ldsAlo[aoRow]),     0, 0);
                __builtin_amdgcn_global_load_async_to_lds_b128(
                    AS1P(&sL[aSrc + 8]), AS3P(&ldsAlo[aoRow + 8]), 0, 0);
#else
                // stage A: contiguous b128 copies, no conversion
                *(v8bf*)&ldsAhi[aoRow]     = *(const v8bf*)&sH[aSrc];
                *(v8bf*)&ldsAhi[aoRow + 8] = *(const v8bf*)&sH[aSrc + 8];
                *(v8bf*)&ldsAlo[aoRow]     = *(const v8bf*)&sL[aSrc];
                *(v8bf*)&ldsAlo[aoRow + 8] = *(const v8bf*)&sL[aSrc + 8];
#endif
                // stage B: transposed bf16 copies to [n][k], zero-padded
#pragma unroll
                for (int j = 0; j < 8; ++j) {
                    const int flat = j * 256 + t;
                    const int k = flat >> 6, n = flat & 63;
                    const int ix = n + kx - 3;
                    __bf16 vh = (__bf16)0.0f, vl = (__bf16)0.0f;
                    if (iy >= 0 && iy < HW && ix >= 0 && ix < HW) {
                        const size_t si = (size_t)(cb + k) * HW * HW + iy * HW + ix;
                        vh = xb[si];
                        vl = xl[si];
                    }
                    ldsBhi[n * LDK + k] = vh;
                    ldsBlo[n * LDK + k] = vl;
                }
#ifdef USE_ASYNC_LDS
                __builtin_amdgcn_s_wait_asynccnt(0);  // own asyncs done pre-barrier
#endif
                __syncthreads();

                // A fragment (ISA 16-bit A 16x32): elems 0-7 = K[8h..8h+8),
                // elems 8-15 = K[16+8h..16+8h+8)
                const int arow = (mo + m16) * LDK;
                const v8bf ah0 = *(const v8bf*)&ldsAhi[arow + 8 * half];
                const v8bf ah1 = *(const v8bf*)&ldsAhi[arow + 16 + 8 * half];
                const v8bf al0 = *(const v8bf*)&ldsAlo[arow + 8 * half];
                const v8bf al1 = *(const v8bf*)&ldsAlo[arow + 16 + 8 * half];
                const v16bf ahi = __builtin_shufflevector(ah0, ah1, 0,1,2,3,4,5,6,7,8,9,10,11,12,13,14,15);
                const v16bf alo = __builtin_shufflevector(al0, al1, 0,1,2,3,4,5,6,7,8,9,10,11,12,13,14,15);
#pragma unroll
                for (int s = 0; s < 4; ++s) {
                    // B fragment (32x16): elems 0-7 = K[8h..8h+8), elems 8-15 = K[16+8h..16+8h+8) at col n
                    const int brow = (s * 16 + m16) * LDK + 8 * half;
                    const v8bf bh0 = *(const v8bf*)&ldsBhi[brow];
                    const v8bf bh1 = *(const v8bf*)&ldsBhi[brow + 16];
                    const v8bf bl0 = *(const v8bf*)&ldsBlo[brow];
                    const v8bf bl1 = *(const v8bf*)&ldsBlo[brow + 16];
                    const v16bf bhi = __builtin_shufflevector(bh0, bh1, 0,1,2,3,4,5,6,7,8,9,10,11,12,13,14,15);
                    const v16bf blo = __builtin_shufflevector(bl0, bl1, 0,1,2,3,4,5,6,7,8,9,10,11,12,13,14,15);
                    tmp[s] = __builtin_amdgcn_wmma_f32_16x16x32_bf16(
                        false, ahi, false, bhi, (short)0, tmp[s], false, false);
                    tmp[s] = __builtin_amdgcn_wmma_f32_16x16x32_bf16(
                        false, ahi, false, blo, (short)0, tmp[s], false, false);
                    tmp[s] = __builtin_amdgcn_wmma_f32_16x16x32_bf16(
                        false, alo, false, bhi, (short)0, tmp[s], false, false);
                }
                __syncthreads();
            }
            // fold the gaussian mask in f32, once per (ky,kx) slab
#pragma unroll
            for (int s = 0; s < 4; ++s)
#pragma unroll
                for (int r = 0; r < 8; ++r)
                    acc[s][r] += mval * tmp[s][r];
        }
    }

    // epilogue: +bias, store conv to d_out, per-channel sum/sumsq via shfl+atomics
#pragma unroll
    for (int r = 0; r < 8; ++r) {
        const int o = mo + r + 8 * half;   // C/D layout: VGPR r -> rows r, r+8
        const float bo = bias[o];
        float psum = 0.0f, psq = 0.0f;
#pragma unroll
        for (int s = 0; s < 4; ++s) {
            const float v = acc[s][r] + bo;
            const int n = s * 16 + m16;
            conv[(((size_t)b * OC + o) * HW + y) * HW + n] = v;
            psum += v;
            psq  += v * v;
        }
#pragma unroll
        for (int msk = 1; msk < 16; msk <<= 1) {
            psum += __shfl_xor(psum, msk, 32);
            psq  += __shfl_xor(psq,  msk, 32);
        }
        if (m16 == 0) {
            atomicAdd(&gsum[o], psum);
            atomicAdd(&gsq[o], psq);
        }
    }
}

// ------------- fold batch stats into per-channel scale/shift ----------------
__global__ void stats_kernel(const float* __restrict__ gsum, const float* __restrict__ gsq,
                             const float* __restrict__ gamma, const float* __restrict__ beta,
                             float* __restrict__ scale, float* __restrict__ shift) {
    const int c = threadIdx.x;
    if (c < OC) {
        const float invN = 1.0f / (float)(BATCH * HW * HW);
        const float mu  = gsum[c] * invN;
        const float var = gsq[c] * invN - mu * mu;
        const float rs  = rsqrtf(var + 1e-5f);
        const float sc  = gamma[c] * rs;
        scale[c] = sc;
        shift[c] = beta[c] - mu * sc;
    }
}

// ------------- in-place BN + exact (erf) GELU -------------------------------
__global__ __launch_bounds__(256) void bn_gelu_kernel(float* __restrict__ out,
                                                      const float* __restrict__ scale,
                                                      const float* __restrict__ shift) {
    const size_t i = (size_t)blockIdx.x * 256 + threadIdx.x;
    const int c = (int)((i >> 12) & (OC - 1));     // /(64*64) % 128
    const float v = out[i] * scale[c] + shift[c];
    out[i] = 0.5f * v * (1.0f + erff(v * 0.70710678118654752f));
}

extern "C" void kernel_launch(void* const* d_in, const int* in_sizes, int n_in,
                              void* d_out, int out_size, void* d_ws, size_t ws_size,
                              hipStream_t stream) {
    const float* x      = (const float*)d_in[0];
    const float* weight = (const float*)d_in[1];
    const float* bias   = (const float*)d_in[2];
    const float* pred_w = (const float*)d_in[3];
    const float* pred_b = (const float*)d_in[4];
    const float* gamma  = (const float*)d_in[5];
    const float* beta   = (const float*)d_in[6];
    float* out = (float*)d_out;

    // ---- workspace carve-up ----
    // [0,16KB): f32 stats block; then bf16 pre-split buffers (16B aligned).
    float* wsf     = (float*)d_ws;
    float* pooled  = wsf;            // 2048 floats
    float* maskbuf = wsf + 2048;     // 784 floats
    float* gsum    = wsf + 2832;     // 128 floats
    float* gsq     = wsf + 2960;     // 128 floats
    float* scale   = wsf + 3088;     // 128 floats
    float* shift   = wsf + 3216;     // 128 floats

    char* wsb = (char*)d_ws + 16384;
    const size_t NW = 49 * 4 * 128 * 32;            // 802816 weight elems
    const size_t NX = (size_t)BATCH * IC * HW * HW; // 8388608 x elems
    __bf16* wtHi = (__bf16*)wsb;                    // 1.53 MB
    __bf16* wtLo = wtHi + NW;                       // 1.53 MB
    __bf16* xHi  = wtLo + NW;                       // 16 MB
    __bf16* xLo  = xHi + NX;                        // 16 MB  (~36.8 MB total)

    (void)hipMemsetAsync(gsum, 0, 256 * sizeof(float), stream);  // gsum+gsq contiguous

    pool_kernel<<<BATCH * IC, 256, 0, stream>>>(x, pooled);
    mask_kernel<<<1, 256, 0, stream>>>(pooled, pred_w, pred_b, maskbuf);
    split_w_kernel<<<(int)(NW / 256), 256, 0, stream>>>(weight, wtHi, wtLo);
    split_x_kernel<<<(int)(NX / 256), 256, 0, stream>>>(x, xHi, xLo);
    conv_wmma_kernel<<<dim3(HW, BATCH), 256, 0, stream>>>(wtHi, wtLo, xHi, xLo,
                                                          bias, maskbuf, out, gsum, gsq);
    stats_kernel<<<1, 128, 0, stream>>>(gsum, gsq, gamma, beta, scale, shift);
    bn_gelu_kernel<<<(int)(NX / 256), 256, 0, stream>>>(out, scale, shift);
}
